// STQuantize_3204045602890
// MI455X (gfx1250) — matrix operand order repaired and checked by reading, not compiled, not run on hardware
//
#include <hip/hip_runtime.h>
#include <math.h>

// ---- problem constants ----
#define KC   512                 // number of codes
#define DC   64                  // code dim / channels
#define NPIX (64*64*64)          // B*H*W = 262144
#define ZQ_N (64*64*64*64)       // 16777216
#define LOSS_OFF  (ZQ_N)
#define IDX_OFF   (ZQ_N + 1)
#define PERP_OFF  (ZQ_N + 1 + NPIX)

#define CB_BYTES  (2048 + 65536)   // enorm + bfrag blob, contiguous in ws

typedef __attribute__((ext_vector_type(16))) __bf16        v16bf;
typedef __attribute__((ext_vector_type(8)))  float         v8f;
typedef __attribute__((ext_vector_type(4)))  unsigned int  v4u;
typedef __attribute__((ext_vector_type(8)))  int           v8i;
typedef __attribute__((ext_vector_type(4)))  int           v4i;

#ifndef __has_builtin
#define __has_builtin(x) 0
#endif

#if __has_builtin(__builtin_amdgcn_tensor_load_to_lds)
#define HAVE_TDM 1
#else
#define HAVE_TDM 0
#endif
#if __has_builtin(__builtin_amdgcn_global_load_async_to_lds_b128)
#define HAVE_ASYNC 1
#else
#define HAVE_ASYNC 0
#endif

// therock-10.0 HIP headers ship the gfx1250 TDM header and the 6-arg builtin;
// ROCm 7.2 (clang-22) has the 5-arg builtin and no header.
#if defined(__has_include)
#if __has_include(<hip/amd_detail/amd_gfx1250_TDM.h>)
#define TDM_SIX_ARGS 1
#else
#define TDM_SIX_ARGS 0
#endif
#else
#define TDM_SIX_ARGS 1
#endif

// workspace layout (bytes):
//   0      : float enorm[512]        (2048)
//   2048   : __bf16 bfrag[32768]     (65536)  -- codebook in WMMA B-fragment order
//   67584  : int   counts[512]       (2048)
//   69632  : float sqerr             (4)

// -------------------------------------------------------------------------
// Kernel A: codebook prep (B-fragment swizzle + ||e||^2) and zero accumulators
// -------------------------------------------------------------------------
__global__ __launch_bounds__(256) void kprep(const float* __restrict__ emb,
                                             float* __restrict__ enorm,
                                             __bf16* __restrict__ bfrag,
                                             int* __restrict__ counts,
                                             float* __restrict__ sqerr) {
    const int tid = threadIdx.x;
    for (int k = tid; k < KC; k += 256) counts[k] = 0;
    if (tid == 0) *sqerr = 0.0f;

    for (int k = tid; k < KC; k += 256) {
        float s = 0.0f;
        #pragma unroll 8
        for (int d = 0; d < DC; ++d) { float e = emb[k * DC + d]; s += e * e; }
        enorm[k] = s;
    }

    // B fragment: tile t (16 codes), K-chunk c (32), lane L, elem j
    //   N = L%16, K = c*32 + (L/16)*16 + j
    for (int e = tid; e < 32 * 2 * 32 * 16; e += 256) {
        const int j = e & 15;
        const int L = (e >> 4) & 31;
        const int c = (e >> 9) & 1;
        const int t = e >> 10;
        const int code = t * 16 + (L & 15);
        const int d    = c * 32 + ((L >> 4) * 16) + j;
        bfrag[e] = (__bf16)emb[code * DC + d];
    }
}

// -------------------------------------------------------------------------
// Kernel B: WMMA distance GEMM + argmin.
// WG = 256 pixels (4 h-rows); each wave: 2 A-tiles (32 pixels) x 512 codes.
// -------------------------------------------------------------------------
__global__ __launch_bounds__(256) void kargmin(const float* __restrict__ z,
                                               const char* __restrict__ cb_g,
                                               float* __restrict__ idx_out) {
    __shared__ __attribute__((aligned(16))) char   s_cb[CB_BYTES];   // enorm + bfrag
    __shared__ __attribute__((aligned(32))) __bf16 s_afrag[16384];   // 32 KB

    const int tid  = threadIdx.x;
    const int lane = tid & 31;
    const int wv   = tid >> 5;

    float*  s_enorm = (float*)s_cb;
    __bf16* s_bfrag = (__bf16*)(s_cb + 2048);

    // ---- stage codebook blob (66 KB) into LDS ----
#if HAVE_TDM
    if (wv == 0) {
        // Tensor DMA: one D# describing a 1-D tile of CB_BYTES/4 dwords.
        const unsigned long long ga = (unsigned long long)cb_g;
        const unsigned int lds_base =
            (unsigned int)(unsigned long long)(__attribute__((address_space(3))) char*)s_cb;
        const unsigned int ndw = CB_BYTES / 4;          // 16896 dwords
        v4u g0;
        g0.x = 1u;                                       // count=1, user mode
        g0.y = lds_base;                                 // lds_addr
        g0.z = (unsigned int)(ga & 0xFFFFFFFFu);         // global_addr[31:0]
        g0.w = (unsigned int)((ga >> 32) & 0x1FFFFFFu)   // global_addr[56:32]
               | (2u << 30);                             // type = 2 ("image")
        v8i g1;
        g1[0] = (int)(2u << 16);                         // data_size = 4 bytes
        g1[1] = (int)((ndw & 0xFFFFu) << 16);            // tensor_dim0[15:0]
        g1[2] = (int)((ndw >> 16) | (1u << 16));         // tensor_dim0[31:16] | tensor_dim1=1
        g1[3] = (int)((ndw & 0xFFFFu) << 16);            // tile_dim0 = ndw
        g1[4] = 0;                                       // tile_dim1/2 unused
        g1[5] = (int)ndw;                                // tensor_dim0_stride
        g1[6] = 0;
        g1[7] = 0;
        const v4i gz4 = {0, 0, 0, 0};
#if TDM_SIX_ARGS
        const v8i gz8 = {0, 0, 0, 0, 0, 0, 0, 0};
        __builtin_amdgcn_tensor_load_to_lds(g0, g1, gz4, gz4, gz8, 0);
#else
        __builtin_amdgcn_tensor_load_to_lds(g0, g1, gz4, gz4, 0);
#endif
#if __has_builtin(__builtin_amdgcn_s_wait_tensorcnt)
        __builtin_amdgcn_s_wait_tensorcnt(0);
#else
        asm volatile("s_wait_tensorcnt 0x0" ::: "memory");
#endif
    }
#elif HAVE_ASYNC
    for (int i = tid; i < CB_BYTES / 16; i += 256) {
        __builtin_amdgcn_global_load_async_to_lds_b128(
            (const __attribute__((address_space(1))) void*)(cb_g + i * 16),
            (__attribute__((address_space(3))) void*)(s_cb + i * 16), 0, 0);
    }
#if __has_builtin(__builtin_amdgcn_s_wait_asynccnt)
    __builtin_amdgcn_s_wait_asynccnt(0);
#else
    asm volatile("s_wait_asynccnt 0x0" ::: "memory");
#endif
#else
    {
        const float4* src = (const float4*)cb_g;
        float4*       dst = (float4*)s_cb;
        #pragma unroll 4
        for (int i = tid; i < CB_BYTES / 16; i += 256) dst[i] = src[i];
    }
#endif

    // ---- load z tile (coalesced across w) and repack into A-fragment bf16 ----
    const int p0 = blockIdx.x * 256;       // first pixel of this WG
    const int b  = p0 >> 12;
    const int h0 = (p0 >> 6) & 63;         // 4 consecutive h rows
    {
        const int wp = tid & 63;           // w position
        const int hl = tid >> 6;           // which h row (0..3)
        const int pixel_local = hl * 64 + wp;       // 0..255
        const int Tt = pixel_local >> 4;            // A-tile index (0..15)
        const int M  = pixel_local & 15;            // row within tile
        const float* zrow = z + ((size_t)(b * 64) * 64 + (h0 + hl)) * 64 + wp;
        #pragma unroll 4
        for (int d = 0; d < DC; ++d) {
            const float v = zrow[(size_t)d * 4096];
            const int c    = d >> 5;
            const int Kloc = d & 31;
            // A layout: K = (L/16)*16 + j  (within 32-chunk)
            const int j   = Kloc & 15;
            const int Lhi = Kloc >> 4;
            const int L   = (Lhi << 4) + M;
            s_afrag[(((Tt * 2 + c) * 32 + L) << 4) + j] = (__bf16)v;
        }
    }
    __syncthreads();

    const int nlane = lane & 15;

    // each wave owns A-tiles u=0,1 -> pixels p0 + wv*32 + u*16 + M
    v16bf a00 = *(const v16bf*)&s_afrag[(((2 * wv + 0) * 2 + 0) * 32 + lane) * 16];
    v16bf a10 = *(const v16bf*)&s_afrag[(((2 * wv + 0) * 2 + 1) * 32 + lane) * 16];
    v16bf a01 = *(const v16bf*)&s_afrag[(((2 * wv + 1) * 2 + 0) * 32 + lane) * 16];
    v16bf a11 = *(const v16bf*)&s_afrag[(((2 * wv + 1) * 2 + 1) * 32 + lane) * 16];

    float best0[8], best1[8]; int bc0[8], bc1[8];
    #pragma unroll
    for (int v = 0; v < 8; ++v) {
        best0[v] = 3.4e38f; best1[v] = 3.4e38f; bc0[v] = 0; bc1[v] = 0;
    }

    // register double-buffering of B fragments across code tiles
    v16bf b0 = *(const v16bf*)&s_bfrag[((0 * 2 + 0) * 32 + lane) * 16];
    v16bf b1 = *(const v16bf*)&s_bfrag[((0 * 2 + 1) * 32 + lane) * 16];

    #pragma unroll 2
    for (int t = 0; t < 32; ++t) {
        const int tn = (t + 1) & 31;                  // wrap: harmless extra load
        v16bf nb0 = *(const v16bf*)&s_bfrag[((tn * 2 + 0) * 32 + lane) * 16];
        v16bf nb1 = *(const v16bf*)&s_bfrag[((tn * 2 + 1) * 32 + lane) * 16];

        v8f acc0 = {}, acc1 = {};
        acc0 = __builtin_amdgcn_wmma_f32_16x16x32_bf16(false, a00, false, b0,
                                                       (short)0, acc0, false, false);
        acc0 = __builtin_amdgcn_wmma_f32_16x16x32_bf16(false, a10, false, b1,
                                                       (short)0, acc0, false, false);
        acc1 = __builtin_amdgcn_wmma_f32_16x16x32_bf16(false, a01, false, b0,
                                                       (short)0, acc1, false, false);
        acc1 = __builtin_amdgcn_wmma_f32_16x16x32_bf16(false, a11, false, b1,
                                                       (short)0, acc1, false, false);

        const int   code = t * 16 + nlane;
        const float en   = s_enorm[code];
        #pragma unroll
        for (int v = 0; v < 8; ++v) {
            const float s0 = en - 2.0f * acc0[v];     // ||e||^2 - 2 z.e
            const float s1 = en - 2.0f * acc1[v];
            if (s0 < best0[v] || (s0 == best0[v] && code < bc0[v])) { best0[v] = s0; bc0[v] = code; }
            if (s1 < best1[v] || (s1 == best1[v] && code < bc1[v])) { best1[v] = s1; bc1[v] = code; }
        }
        b0 = nb0; b1 = nb1;
    }

    // butterfly argmin across the 16 lanes holding different codes (per half)
    #pragma unroll
    for (int off = 8; off > 0; off >>= 1) {
        #pragma unroll
        for (int v = 0; v < 8; ++v) {
            float s2; int c2;
            s2 = __shfl_xor(best0[v], off, 32);
            c2 = __shfl_xor(bc0[v],  off, 32);
            if (s2 < best0[v] || (s2 == best0[v] && c2 < bc0[v])) { best0[v] = s2; bc0[v] = c2; }
            s2 = __shfl_xor(best1[v], off, 32);
            c2 = __shfl_xor(bc1[v],  off, 32);
            if (s2 < best1[v] || (s2 == best1[v] && c2 < bc1[v])) { best1[v] = s2; bc1[v] = c2; }
        }
    }

    if (nlane == 0) {                    // lanes 0 and 16 hold M=0..7 / 8..15
        const int Mbase = (lane >> 4) * 8;
        #pragma unroll
        for (int v = 0; v < 8; ++v) {
            idx_out[p0 + wv * 32 +  0 + Mbase + v] = (float)bc0[v];
            idx_out[p0 + wv * 32 + 16 + Mbase + v] = (float)bc1[v];
        }
    }
}

// -------------------------------------------------------------------------
// Kernel C: gather z_q (B,C,H,W), fp32 MSE partial sums, code histogram
// -------------------------------------------------------------------------
__global__ __launch_bounds__(256) void kgather(const float* __restrict__ z,
                                               const float* __restrict__ emb,
                                               const float* __restrict__ idx_f,
                                               float* __restrict__ zq_out,
                                               int* __restrict__ counts,
                                               float* __restrict__ sqerr) {
    __shared__ int hist[KC];
    for (int k = threadIdx.x; k < KC; k += 256) hist[k] = 0;
    __syncthreads();

    const int p  = blockIdx.x * 256 + threadIdx.x;
    const int wp = p & 63, h = (p >> 6) & 63, b = p >> 12;
    const int k  = (int)idx_f[p];
    const float* e = emb + k * DC;
    const size_t base = ((size_t)b * 64 * 64 * 64) + (size_t)h * 64 + wp;

    float sq = 0.0f;
    #pragma unroll 8
    for (int c = 0; c < DC; ++c) {
        const float  ev  = e[c];
        const size_t off = base + (size_t)c * 4096;
        const float  d   = ev - z[off];
        sq += d * d;
        zq_out[off] = ev;               // forward value of STE output == z_q
    }
    atomicAdd(&hist[k], 1);

    #pragma unroll
    for (int off = 16; off > 0; off >>= 1) sq += __shfl_xor(sq, off, 32);
    if ((threadIdx.x & 31) == 0) atomicAdd(sqerr, sq);

    __syncthreads();
    for (int kk = threadIdx.x; kk < KC; kk += 256)
        atomicAdd(&counts[kk], hist[kk]);
}

// -------------------------------------------------------------------------
// Kernel D: loss + perplexity scalars
// -------------------------------------------------------------------------
__global__ __launch_bounds__(256) void kfinal(const int* __restrict__ counts,
                                              const float* __restrict__ sqerr,
                                              float* __restrict__ d_out) {
    __shared__ float part[256];
    float s = 0.0f;
    for (int k = threadIdx.x; k < KC; k += 256) {
        const float avg = (float)counts[k] / (float)NPIX;
        s += avg * logf(avg + 1e-10f);
    }
    part[threadIdx.x] = s;
    __syncthreads();
    for (int off = 128; off > 0; off >>= 1) {
        if (threadIdx.x < off) part[threadIdx.x] += part[threadIdx.x + off];
        __syncthreads();
    }
    if (threadIdx.x == 0) {
        d_out[PERP_OFF] = expf(-part[0]);
        d_out[LOSS_OFF] = 1.25f * (*sqerr) / (float)ZQ_N;  // (1 + 0.25) * MSE
    }
}

// -------------------------------------------------------------------------
extern "C" void kernel_launch(void* const* d_in, const int* in_sizes, int n_in,
                              void* d_out, int out_size, void* d_ws, size_t ws_size,
                              hipStream_t stream) {
    (void)in_sizes; (void)n_in; (void)out_size; (void)ws_size;
    const float* z   = (const float*)d_in[0];
    const float* emb = (const float*)d_in[1];
    float* out = (float*)d_out;

    char*   ws     = (char*)d_ws;
    float*  enorm  = (float*)(ws);
    __bf16* bfrag  = (__bf16*)(ws + 2048);
    int*    counts = (int*)(ws + CB_BYTES);
    float*  sqerr  = (float*)(ws + CB_BYTES + 2048);

    kprep  <<<1,          256, 0, stream>>>(emb, enorm, bfrag, counts, sqerr);
    kargmin<<<NPIX / 256, 256, 0, stream>>>(z, ws, out + IDX_OFF);
    kgather<<<NPIX / 256, 256, 0, stream>>>(z, emb, out + IDX_OFF, out, counts, sqerr);
    kfinal <<<1,          256, 0, stream>>>(counts, sqerr, out);
}
